// AdditiveAttention_52845277610075
// MI455X (gfx1250) — compile-verified, matrix operand order checked by reading, not run on hardware
//
#include <hip/hip_runtime.h>
#include <hip/hip_bf16.h>

typedef __attribute__((ext_vector_type(16))) __bf16 v16bf;
typedef __attribute__((ext_vector_type(8)))  float  v8f;

constexpr int Bn = 32;
constexpr int Tn = 2048;
constexpr int Hn = 1024;
constexpr int NTILES  = Hn / 16;  // 64 N-tiles of 16 output columns
constexpr int KSTEPS  = Hn / 32;  // 32 K-steps of 32 (bf16 WMMA depth)
constexpr int NTBLOCK = 4;        // N-tiles register-blocked per wave

// ---- native tanh (V_TANH_F32) ---------------------------------------------
#if defined(__has_builtin)
#if __has_builtin(__builtin_amdgcn_tanhf)
#define TANHF(x) __builtin_amdgcn_tanhf(x)
#endif
#if __has_builtin(__builtin_amdgcn_cvt_pk_bf16_f32)
#define HAVE_CVT_PK_BF16 1
#endif
#endif
#ifndef TANHF
__device__ __forceinline__ float dev_tanh(float x) {
    float y;
    asm volatile("v_tanh_f32 %0, %1\n\tv_nop" : "=v"(y) : "v"(x));
    return y;
}
#define TANHF(x) dev_tanh(x)
#endif

// ---- pack two f32 into one bf16x2 dword: native V_CVT_PK_BF16_F32 ----------
__device__ __forceinline__ unsigned pk2bf(float a, float b) {
#ifdef HAVE_CVT_PK_BF16
    auto p = __builtin_amdgcn_cvt_pk_bf16_f32(a, b);
    static_assert(sizeof(p) == 4, "unexpected cvt_pk_bf16 return size");
    unsigned d;
    __builtin_memcpy(&d, &p, 4);
    return d;
#else
    unsigned d;
    asm("v_cvt_pk_bf16_f32 %0, %1, %2" : "=v"(d) : "v"(a), "v"(b));
    return d;
#endif
}

// ---------------------------------------------------------------------------
// Pack Wk (f32 [H][H]) into per-lane bf16 B-operand fragments for
// v_wmma_f32_16x16x32_bf16. Per (ntile,kstep): 32 lanes x 8 dwords contiguous.
// lane l: N = ntile*16 + (l&15), khalf = l>>4
//   dwords 0..3 -> K = k0+8*khalf + {0..7}
//   dwords 4..7 -> K = k0+16+8*khalf + {0..7}
// ---------------------------------------------------------------------------
__global__ void packwk_kernel(const float* __restrict__ Wk,
                              unsigned* __restrict__ WkP) {
    const int ks   = blockIdx.x;
    const int nt   = blockIdx.y;
    const int lane = threadIdx.x;
    const int l15   = lane & 15;
    const int khalf = lane >> 4;
    const int n  = nt * 16 + l15;
    const int k0 = ks * 32 + khalf * 8;
    const float* row = Wk + (size_t)n * Hn;
    unsigned* dst = WkP + ((((size_t)nt * KSTEPS + ks) * 32) + lane) * 8;
#pragma unroll
    for (int j = 0; j < 4; ++j)
        dst[j] = pk2bf(row[k0 + 2 * j], row[k0 + 2 * j + 1]);
#pragma unroll
    for (int j = 0; j < 4; ++j)
        dst[4 + j] = pk2bf(row[k0 + 16 + 2 * j], row[k0 + 16 + 2 * j + 1]);
}

// ---------------------------------------------------------------------------
// qq[b][h] = sum_j dec[b][j]*Wq[h][j] + bq[h] + bk[h]   (tiny GEMV)
// ---------------------------------------------------------------------------
__global__ void qk_kernel(const float* __restrict__ dec,
                          const float* __restrict__ Wq,
                          const float* __restrict__ bq,
                          const float* __restrict__ bk,
                          float* __restrict__ qq) {
    __shared__ float ds[Hn];
    const int b = blockIdx.y;
    const int h = blockIdx.x * 256 + threadIdx.x;
    for (int j = threadIdx.x; j < Hn; j += 256) ds[j] = dec[b * Hn + j];
    __syncthreads();
    const float* wrow = Wq + (size_t)h * Hn;
    float acc = 0.f;
    for (int j = 0; j < Hn; ++j) acc = fmaf(ds[j], wrow[j], acc);
    qq[b * Hn + h] = acc + bq[h] + bk[h];
}

// ---------------------------------------------------------------------------
// Fused: score[b][t] = tanh(enc[b]·Wk^T + qq[b])·Wout + bout
// 8 waves/block; wave owns a 16-row t-strip. Register-blocks 4 N-tiles:
// per K-step one A load+convert feeds 4 v_wmma_f32_16x16x32_bf16.
// ---------------------------------------------------------------------------
__global__ __launch_bounds__(256)
void score_kernel(const float* __restrict__ enc,
                  const unsigned* __restrict__ WkP,
                  const float* __restrict__ qq,
                  const float* __restrict__ Wout,
                  const float* __restrict__ bout,
                  float* __restrict__ score) {
    const int lane  = threadIdx.x & 31;
    const int wave  = threadIdx.x >> 5;
    const int b     = blockIdx.y;
    const int t0    = blockIdx.x * 128 + wave * 16;
    const int l15   = lane & 15;
    const int khalf = lane >> 4;
    const int m     = t0 + l15;                       // A-matrix row (t index)
    const float* encRow = enc + ((size_t)b * Tn + m) * Hn;

    float s[8];
#pragma unroll
    for (int r = 0; r < 8; ++r) s[r] = 0.f;

    for (int ntb = 0; ntb < NTILES / NTBLOCK; ++ntb) {
        float qv[NTBLOCK], wv[NTBLOCK];
#pragma unroll
        for (int j = 0; j < NTBLOCK; ++j) {
            const int n = (ntb * NTBLOCK + j) * 16 + l15;
            qv[j] = qq[b * Hn + n];
            wv[j] = Wout[n];
        }
        const unsigned* bbase =
            WkP + ((size_t)(ntb * NTBLOCK) * KSTEPS * 32) * 8;

        v8f acc[NTBLOCK];
#pragma unroll
        for (int j = 0; j < NTBLOCK; ++j) acc[j] = (v8f){};

#pragma unroll 2
        for (int ks = 0; ks < KSTEPS; ++ks) {
            const int k0 = ks * 32 + khalf * 8;
            float4 f0 = *(const float4*)(encRow + k0);
            float4 f1 = *(const float4*)(encRow + k0 + 4);
            float4 f2 = *(const float4*)(encRow + k0 + 16);
            float4 f3 = *(const float4*)(encRow + k0 + 20);
            __builtin_prefetch(encRow + k0 + 32, 0, 3);  // near-cache prefetch
            union { v16bf v; unsigned d[8]; } A;
            A.d[0] = pk2bf(f0.x, f0.y);
            A.d[1] = pk2bf(f0.z, f0.w);
            A.d[2] = pk2bf(f1.x, f1.y);
            A.d[3] = pk2bf(f1.z, f1.w);
            A.d[4] = pk2bf(f2.x, f2.y);
            A.d[5] = pk2bf(f2.z, f2.w);
            A.d[6] = pk2bf(f3.x, f3.y);
            A.d[7] = pk2bf(f3.z, f3.w);
#pragma unroll
            for (int j = 0; j < NTBLOCK; ++j) {
                const v16bf Bf = *(const v16bf*)(
                    bbase + ((size_t)j * KSTEPS * 32 + (size_t)ks * 32 + lane) * 8);
                acc[j] = __builtin_amdgcn_wmma_f32_16x16x32_bf16(
                             false, A.v, false, Bf, (short)0, acc[j],
                             false, false);
            }
        }
        // epilogue: fold this N-block into per-row score partials
#pragma unroll
        for (int j = 0; j < NTBLOCK; ++j)
#pragma unroll
            for (int r = 0; r < 8; ++r)
                s[r] += TANHF(acc[j][r] + qv[j]) * wv[j];
    }

    const float bo = bout[0];
#pragma unroll
    for (int r = 0; r < 8; ++r) {
        float v = s[r];
        v += __shfl_xor(v, 1, 16);
        v += __shfl_xor(v, 2, 16);
        v += __shfl_xor(v, 4, 16);
        v += __shfl_xor(v, 8, 16);
        if (l15 == 0)   // lanes 0-15 -> rows t0+0..7 ; lanes 16-31 -> t0+8..15
            score[(size_t)b * Tn + t0 + khalf * 8 + r] = v + bo;
    }
}

// ---------------------------------------------------------------------------
// softmax over T per batch row
// ---------------------------------------------------------------------------
__global__ void softmax_kernel(const float* __restrict__ score,
                               float* __restrict__ attn) {
    __shared__ float red[256];
    const int b = blockIdx.x, tid = threadIdx.x;
    const float* srow = score + (size_t)b * Tn;
    float mx = -3.4e38f;
    for (int t = tid; t < Tn; t += 256) mx = fmaxf(mx, srow[t]);
    red[tid] = mx; __syncthreads();
    for (int sft = 128; sft > 0; sft >>= 1) {
        if (tid < sft) red[tid] = fmaxf(red[tid], red[tid + sft]);
        __syncthreads();
    }
    mx = red[0]; __syncthreads();
    float sum = 0.f;
    for (int t = tid; t < Tn; t += 256) sum += __expf(srow[t] - mx);
    red[tid] = sum; __syncthreads();
    for (int sft = 128; sft > 0; sft >>= 1) {
        if (tid < sft) red[tid] += red[tid + sft];
        __syncthreads();
    }
    const float inv = 1.0f / red[0];
    for (int t = tid; t < Tn; t += 256)
        attn[(size_t)b * Tn + t] = __expf(srow[t] - mx) * inv;
}

// ---------------------------------------------------------------------------
// context partials + reduction
// ---------------------------------------------------------------------------
__global__ void ctx_kernel(const float* __restrict__ enc,
                           const float* __restrict__ attn,
                           float* __restrict__ partial) {
    const int c = blockIdx.x, b = blockIdx.y;
    const int h0 = threadIdx.x * 4;
    float4 acc = {0.f, 0.f, 0.f, 0.f};
    const int tBeg = c * (Tn / 8), tEnd = tBeg + (Tn / 8);
    for (int t = tBeg; t < tEnd; ++t) {
        const float a = attn[(size_t)b * Tn + t];
        float4 e = *(const float4*)(enc + ((size_t)b * Tn + t) * Hn + h0);
        acc.x = fmaf(a, e.x, acc.x);
        acc.y = fmaf(a, e.y, acc.y);
        acc.z = fmaf(a, e.z, acc.z);
        acc.w = fmaf(a, e.w, acc.w);
    }
    *(float4*)(partial + ((size_t)(b * 8 + c)) * Hn + h0) = acc;
}

__global__ void ctxred_kernel(const float* __restrict__ partial,
                              float* __restrict__ out) {
    const int i = blockIdx.x * 256 + threadIdx.x;   // i in [0, B*H)
    const int b = i >> 10, h = i & (Hn - 1);
    float sum = 0.f;
#pragma unroll
    for (int c = 0; c < 8; ++c) sum += partial[((size_t)(b * 8 + c)) * Hn + h];
    out[i] = sum;
}

// ---------------------------------------------------------------------------
extern "C" void kernel_launch(void* const* d_in, const int* in_sizes, int n_in,
                              void* d_out, int out_size, void* d_ws, size_t ws_size,
                              hipStream_t stream) {
    const float* enc  = (const float*)d_in[0];   // [B,T,H]
    const float* dec  = (const float*)d_in[1];   // [1,B,H]
    const float* Wk   = (const float*)d_in[2];   // [H,H]
    const float* bk   = (const float*)d_in[3];   // [H]
    const float* Wq   = (const float*)d_in[4];   // [H,H]
    const float* bq   = (const float*)d_in[5];   // [H]
    const float* Wout = (const float*)d_in[6];   // [H]
    const float* bout = (const float*)d_in[7];   // scalar
    float* out = (float*)d_out;                  // [B,1,H]

    char* ws = (char*)d_ws;
    unsigned* WkP  = (unsigned*)(ws);                                  // 2 MiB
    float* qq      = (float*)(ws + (size_t)2 * 1024 * 1024);           // 128 KiB
    float* score   = (float*)(ws + (size_t)2 * 1024 * 1024 + 131072);  // 256 KiB
    float* attn    = (float*)(ws + (size_t)2 * 1024 * 1024 + 131072 + 262144);
    float* partial = (float*)(ws + (size_t)2 * 1024 * 1024 + 131072 + 262144 + 262144);

    packwk_kernel <<<dim3(KSTEPS, NTILES), 32, 0, stream>>>(Wk, WkP);
    qk_kernel     <<<dim3(Hn / 256, Bn), 256, 0, stream>>>(dec, Wq, bq, bk, qq);
    score_kernel  <<<dim3(Tn / 128, Bn), 256, 0, stream>>>(enc, WkP, qq, Wout, bout, score);
    softmax_kernel<<<Bn, 256, 0, stream>>>(score, attn);
    ctx_kernel    <<<dim3(8, Bn), 256, 0, stream>>>(enc, attn, partial);
    ctxred_kernel <<<(Bn * Hn) / 256, 256, 0, stream>>>(partial, out);
}
